// GraphConvLayer_45672682226314
// MI455X (gfx1250) — matrix operand order repaired and checked by reading, not compiled, not run on hardware
//
#include <hip/hip_runtime.h>
#include <hip/hip_bf16.h>
#include <math.h>

// Problem sizes (fixed by the reference)
#define BB 32
#define NN 1024
#define DD 256

typedef __attribute__((ext_vector_type(16))) __bf16       v16bf;
typedef __attribute__((ext_vector_type(8)))  float        v8f;
typedef __attribute__((ext_vector_type(4)))  unsigned int v4u;
typedef __attribute__((ext_vector_type(4)))  unsigned int u32x4;
typedef __attribute__((ext_vector_type(8)))  int          i32x8;
typedef __attribute__((ext_vector_type(4)))  int          i32x4;

union Frag { v16bf bf; v4u q[2]; };

__device__ __forceinline__ unsigned packbf(float a, float b) {
  unsigned ua = __float_as_uint(a), ub = __float_as_uint(b);
  ua = (ua + 0x7FFFu + ((ua >> 16) & 1u)) >> 16;   // round-to-nearest-even
  ub = (ub + 0x7FFFu + ((ub >> 16) & 1u)) >> 16;
  return (ua & 0xFFFFu) | (ub << 16);
}

__device__ __forceinline__ float gelu_exact(float x) {
  return 0.5f * x * (1.0f + erff(x * 0.70710678118654752f));
}

// ---------------------------------------------------------------------------
// Tensor Data Mover: load a 64-row x 128-col bf16 tile of adj_n into LDS.
// D# per CDNA5 ISA:
//  group0: [1:0]=count=1, [63:32]=lds_addr, [120:64]=global_addr, [127:126]=2
//  group1: data_size=2B, tensor_dim0=1024, tensor_dim1=1024, tile_dim0=128,
//          tile_dim1=64, tensor_dim0_stride=1024 (elements)
// ---------------------------------------------------------------------------
__device__ __forceinline__ void tdm_load_tile(unsigned lds_addr,
                                              unsigned long long gaddr) {
  u32x4 g0;
  g0.x = 1u;                                        // count = 1, user mode
  g0.y = lds_addr;                                  // LDS byte address
  g0.z = (unsigned)(gaddr & 0xFFFFFFFFull);         // global_addr[31:0]
  g0.w = (unsigned)(gaddr >> 32) | 0x80000000u;     // [56:32] | type=2
  i32x8 g1 = { (int)(1u << 16),                     // data_size = 1 -> 2 bytes
               (int)(1024u << 16),                  // tensor_dim0[15:0]=1024
               (int)(1024u << 16),                  // tensor_dim1[15:0]=1024
               (int)(128u << 16),                   // tile_dim0 = 128 elems
               (int)64,                             // tile_dim1 = 64 rows
               (int)1024,                           // tensor_dim0_stride
               0, 0 };
  i32x4 z4 = {0, 0, 0, 0};
#if defined(__clang_major__) && (__clang_major__ >= 23)
  i32x8 z8 = {0, 0, 0, 0, 0, 0, 0, 0};
  __builtin_amdgcn_tensor_load_to_lds(g0, g1, z4, z4, z8, 0);
#else
  __builtin_amdgcn_tensor_load_to_lds(g0, g1, z4, z4, 0);
#endif
}

// ---------------------------------------------------------------------------
// P1: per-row degree reduction fused with adj -> bf16 row-scaled conversion.
// ---------------------------------------------------------------------------
__global__ void __launch_bounds__(256) prep_adj_kernel(
    const float* __restrict__ adj, char* __restrict__ adjn) {
  __shared__ float red[256];
  const int row = blockIdx.x;                    // b*NN + n, 0..32767
  const int t = threadIdx.x;
  const float4 v = ((const float4*)(adj + (size_t)row * NN))[t];
  red[t] = v.x + v.y + v.z + v.w;
  __syncthreads();
  for (int off = 128; off > 0; off >>= 1) {
    if (t < off) red[t] += red[t + off];
    __syncthreads();
  }
  float s = red[0];
  float dinv = (s != 0.0f) ? (1.0f / s) : 0.0f;
  uint2 o;
  o.x = packbf(v.x * dinv, v.y * dinv);
  o.y = packbf(v.z * dinv, v.w * dinv);
  ((uint2*)(adjn + (size_t)row * (NN * 2)))[t] = o;
}

// ---------------------------------------------------------------------------
// P2: x fp32 -> bf16 row-major (A-operand form).
// ---------------------------------------------------------------------------
__global__ void __launch_bounds__(256) prep_x_kernel(
    const float* __restrict__ x, char* __restrict__ xrm) {
  const size_t i = (size_t)blockIdx.x * 256 + threadIdx.x;  // one float4 each
  const float4 v = ((const float4*)x)[i];
  uint2 o;
  o.x = packbf(v.x, v.y);
  o.y = packbf(v.z, v.w);
  ((uint2*)xrm)[i] = o;
}

// ---------------------------------------------------------------------------
// P3: build W^T in B-fragment-swizzled bf16 layout for all 3 layers.
// ---------------------------------------------------------------------------
__global__ void __launch_bounds__(256) prep_w_kernel(
    const float* __restrict__ W0, const float* __restrict__ W1,
    const float* __restrict__ W2, char* __restrict__ wswz) {
  const int id = blockIdx.x * 256 + threadIdx.x;  // 0..12287
  const int layer = id >> 12;                     // 8*16*32 = 4096 per layer
  const int kt = (id >> 9) & 7;
  const int nt = (id >> 5) & 15;
  const int l = id & 31;
  const int lm = l & 15, half = l >> 4;
  const float* W = (layer == 0) ? W0 : (layer == 1) ? W1 : W2;
  const int n = nt * 16 + lm;
  const int kb = kt * 32 + half * 16;
  const float* src = W + (size_t)n * DD + kb;     // 16 contiguous floats
  v4u o0, o1;
  o0.x = packbf(src[0], src[1]);   o0.y = packbf(src[2], src[3]);
  o0.z = packbf(src[4], src[5]);   o0.w = packbf(src[6], src[7]);
  o1.x = packbf(src[8], src[9]);   o1.y = packbf(src[10], src[11]);
  o1.z = packbf(src[12], src[13]); o1.w = packbf(src[14], src[15]);
  char* dst = wswz + (size_t)id * 32;
  ((v4u*)dst)[0] = o0;
  ((v4u*)dst)[1] = o1;
}

// ---------------------------------------------------------------------------
// Linear: y = x @ W^T + b.  A = x (bf16 row-major), B = W^T (swizzled).
// Loads all 4 B fragments into distinct registers before the WMMA burst.
// ---------------------------------------------------------------------------
__global__ void __launch_bounds__(256) lin_kernel(
    const char* __restrict__ xrm, const char* __restrict__ wswz,
    const float* __restrict__ bias, char* __restrict__ yswz) {
  const int b = blockIdx.x >> 5;
  const int mblk = blockIdx.x & 31;
  const int tid = threadIdx.x;
  const int wave = tid >> 5, l = tid & 31, lm = l & 15, half = l >> 4;
  const int mt = mblk * 2 + (wave >> 2);    // 0..63
  const int ntg = wave & 3;
  const int row = mt * 16 + lm;

  const char* abase = xrm + (((size_t)b * NN + row) * DD) * 2 + half * 32;
  v8f acc[4];
#pragma unroll
  for (int j = 0; j < 4; ++j) acc[j] = (v8f){0, 0, 0, 0, 0, 0, 0, 0};

#pragma unroll
  for (int kt = 0; kt < 8; ++kt) {
    Frag A;
    A.q[0] = *(const v4u*)(abase + kt * 64);
    A.q[1] = *(const v4u*)(abase + kt * 64 + 16);
    Frag Bf[4];
#pragma unroll
    for (int j = 0; j < 4; ++j) {
      const char* bp = wswz + (((size_t)kt * 16 + ntg * 4 + j) * 32 + l) * 32;
      Bf[j].q[0] = *(const v4u*)(bp);
      Bf[j].q[1] = *(const v4u*)(bp + 16);
    }
#pragma unroll
    for (int j = 0; j < 4; ++j)
      acc[j] = __builtin_amdgcn_wmma_f32_16x16x32_bf16(
          false, A.bf, false, Bf[j].bf, (short)0, acc[j], false, false);
  }

  const int ktY = mt >> 1;
#pragma unroll
  for (int j = 0; j < 4; ++j) {
    const int nt = ntg * 4 + j;
    const float bv = bias[nt * 16 + lm];
    v4u o;
    o.x = packbf(acc[j][0] + bv, acc[j][1] + bv);
    o.y = packbf(acc[j][2] + bv, acc[j][3] + bv);
    o.z = packbf(acc[j][4] + bv, acc[j][5] + bv);
    o.w = packbf(acc[j][6] + bv, acc[j][7] + bv);
    char* dst = yswz + (size_t)b * (NN * DD * 2)
              + ((((size_t)ktY * 16 + nt) * 32) + lm + 16 * (mt & 1)) * 32
              + half * 16;
    *(v4u*)dst = o;
  }
}

// ---------------------------------------------------------------------------
// Aggregation: out = gelu(mask * (adj_n @ y)).
// TDM double-buffers 64x128 bf16 adj tiles into LDS (wave 0 issues, waits
// s_wait_tensorcnt, block barrier publishes).  Each wave owns 2 M-tiles x
// 4 N-tiles (8 accumulators); A fragments from LDS, B streamed from L2 with
// prefetch.  Epilog transposes through the same LDS pool in two passes.
// grid: 32 batches * 16 row-blocks (64 rows each); block: 256 threads.
// ---------------------------------------------------------------------------
__global__ void __launch_bounds__(256) agg_kernel(
    const char* __restrict__ adjn, const char* __restrict__ yswz,
    const float* __restrict__ mask, char* __restrict__ xrm_out,
    float* __restrict__ fout, int final_layer) {
  __shared__ __align__(1024) char pool[32768];    // 2x16KB bufs / 32KB epilog
  const int b = blockIdx.x >> 4;
  const int mblk = blockIdx.x & 15;
  const int rowbase = mblk * 64;
  const int tid = threadIdx.x;
  const int wave = tid >> 5, l = tid & 31, lm = l & 15, half = l >> 4;
  const int mtp = (wave >> 2) * 2;                // wave's first local M-tile
  const int ntg = wave & 3;

  const char* bb = yswz + (size_t)b * (NN * DD * 2);
  const unsigned long long abase = (unsigned long long)(uintptr_t)adjn
      + ((unsigned long long)b * NN + (unsigned long long)rowbase) * (NN * 2);
  const unsigned lds_base = (unsigned)(uintptr_t)(void*)pool;

  v8f acc[2][4];
#pragma unroll
  for (int p = 0; p < 2; ++p)
#pragma unroll
    for (int j = 0; j < 4; ++j) acc[p][j] = (v8f){0, 0, 0, 0, 0, 0, 0, 0};

  if (wave == 0) tdm_load_tile(lds_base, abase);           // chunk 0

  for (int c = 0; c < 8; ++c) {                            // 8 x 128-wide K
    if (wave == 0) {
      if (c < 7) {
        tdm_load_tile(lds_base + (((c + 1) & 1) ? 16384u : 0u),
                      abase + (unsigned long long)(c + 1) * 256ull);
        __builtin_amdgcn_s_wait_tensorcnt(1);              // chunk c landed
      } else {
        __builtin_amdgcn_s_wait_tensorcnt(0);
      }
    }
    __syncthreads();                                       // publish chunk c
    const char* buf = pool + ((c & 1) ? 16384 : 0);

    if (c < 7) {                                           // prefetch next B
#pragma unroll
      for (int ktl = 0; ktl < 4; ++ktl)
        __builtin_prefetch(
            bb + ((((size_t)(c + 1) * 4 + ktl) * 16 + ntg * 4) * 32 + l) * 32,
            0, 1);
    }

#pragma unroll
    for (int ktl = 0; ktl < 4; ++ktl) {
      const int kt = c * 4 + ktl;
      Frag A0, A1;
      const char* a0 = buf + (mtp * 16 + lm) * 256 + ktl * 64 + half * 32;
      A0.q[0] = *(const v4u*)(a0);
      A0.q[1] = *(const v4u*)(a0 + 16);
      const char* a1 = a0 + 16 * 256;
      A1.q[0] = *(const v4u*)(a1);
      A1.q[1] = *(const v4u*)(a1 + 16);
      Frag Bf[4];
#pragma unroll
      for (int j = 0; j < 4; ++j) {
        const char* bp = bb + (((size_t)kt * 16 + ntg * 4 + j) * 32 + l) * 32;
        Bf[j].q[0] = *(const v4u*)(bp);
        Bf[j].q[1] = *(const v4u*)(bp + 16);
      }
#pragma unroll
      for (int j = 0; j < 4; ++j) {
        acc[0][j] = __builtin_amdgcn_wmma_f32_16x16x32_bf16(
            false, A0.bf, false, Bf[j].bf, (short)0, acc[0][j], false, false);
        acc[1][j] = __builtin_amdgcn_wmma_f32_16x16x32_bf16(
            false, A1.bf, false, Bf[j].bf, (short)0, acc[1][j], false, false);
      }
    }
    __syncthreads();                         // done reading buf before reuse
  }

  // --- epilog: transpose D-tiles via LDS, one M-tile pass at a time ---
  float* wl = (float*)pool + wave * 1024;    // 16 rows x 64 cols per wave
#pragma unroll
  for (int p = 0; p < 2; ++p) {
    __syncthreads();
#pragma unroll
    for (int j = 0; j < 4; ++j)
#pragma unroll
      for (int v = 0; v < 8; ++v)
        wl[(v + half * 8) * 64 + j * 16 + lm] = acc[p][j][v];
    __syncthreads();

    const int m = rowbase + (mtp + p) * 16 + lm;
    const float mk = mask[b * NN + m];
    const float4* rdp = (const float4*)(wl + lm * 64 + half * 32);
    float vals[32];
#pragma unroll
    for (int i = 0; i < 8; ++i) {
      const float4 vv = rdp[i];
      vals[4 * i + 0] = gelu_exact(vv.x * mk);
      vals[4 * i + 1] = gelu_exact(vv.y * mk);
      vals[4 * i + 2] = gelu_exact(vv.z * mk);
      vals[4 * i + 3] = gelu_exact(vv.w * mk);
    }
    const int colbase = ntg * 64 + half * 32;

    if (final_layer) {
      float* dst = fout + ((size_t)b * NN + m) * DD + colbase;
#pragma unroll
      for (int i = 0; i < 8; ++i)
        ((float4*)dst)[i] = make_float4(vals[4 * i], vals[4 * i + 1],
                                        vals[4 * i + 2], vals[4 * i + 3]);
    } else {
      char* dst = xrm_out + (((size_t)b * NN + m) * DD + colbase) * 2;
#pragma unroll
      for (int i = 0; i < 4; ++i) {
        v4u o;
        o.x = packbf(vals[8 * i + 0], vals[8 * i + 1]);
        o.y = packbf(vals[8 * i + 2], vals[8 * i + 3]);
        o.z = packbf(vals[8 * i + 4], vals[8 * i + 5]);
        o.w = packbf(vals[8 * i + 6], vals[8 * i + 7]);
        ((v4u*)dst)[i] = o;
      }
    }
  }
}

// ---------------------------------------------------------------------------
// Workspace layout (bytes):
//   [0, 64MB)        adj_n bf16 row-major (row-scaled), L2-resident
//   [64MB, 80MB)     x_rm  bf16 row-major activations (A operand)
//   [80MB, 96MB)     y_swz bf16 B-fragment-swizzled activations
//   [96MB, +384KB)   w_swz bf16 W^T swizzled, 3 layers
// ---------------------------------------------------------------------------
extern "C" void kernel_launch(void* const* d_in, const int* in_sizes, int n_in,
                              void* d_out, int out_size, void* d_ws,
                              size_t ws_size, hipStream_t stream) {
  (void)in_sizes; (void)n_in; (void)out_size; (void)ws_size;
  const float* x    = (const float*)d_in[0];
  const float* mask = (const float*)d_in[1];
  const float* adj  = (const float*)d_in[2];
  const float* W0   = (const float*)d_in[3];
  const float* b0   = (const float*)d_in[4];
  const float* W1   = (const float*)d_in[5];
  const float* b1   = (const float*)d_in[6];
  const float* W2   = (const float*)d_in[7];
  const float* b2   = (const float*)d_in[8];
  float* out = (float*)d_out;

  char* ws = (char*)d_ws;
  char* adjn = ws;                                  // 64 MB
  char* xrm  = ws + (size_t)64 * 1024 * 1024;       // 16 MB
  char* yswz = ws + (size_t)80 * 1024 * 1024;       // 16 MB
  char* wswz = ws + (size_t)96 * 1024 * 1024;       // 384 KB

  prep_adj_kernel<<<BB * NN, 256, 0, stream>>>(adj, adjn);
  prep_x_kernel<<<(BB * NN * DD) / 4 / 256, 256, 0, stream>>>(x, xrm);
  prep_w_kernel<<<48, 256, 0, stream>>>(W0, W1, W2, wswz);

  const int lin_grid = BB * (NN / 32);              // 1024 blocks
  const int agg_grid = BB * (NN / 64);              // 512 blocks
  const size_t wstride = (size_t)4096 * 32;         // 128 KB per layer

  lin_kernel<<<lin_grid, 256, 0, stream>>>(xrm, wswz + 0 * wstride, b0, yswz);
  agg_kernel<<<agg_grid, 256, 0, stream>>>(adjn, yswz, mask, xrm, out, 0);
  lin_kernel<<<lin_grid, 256, 0, stream>>>(xrm, wswz + 1 * wstride, b1, yswz);
  agg_kernel<<<agg_grid, 256, 0, stream>>>(adjn, yswz, mask, xrm, out, 0);
  lin_kernel<<<lin_grid, 256, 0, stream>>>(xrm, wswz + 2 * wstride, b2, yswz);
  agg_kernel<<<agg_grid, 256, 0, stream>>>(adjn, yswz, mask, xrm, out, 1);
}
